// AcousticEncoder_87823491268903
// MI455X (gfx1250) — matrix operand order repaired and checked by reading, not compiled, not run on hardware
//
#include <hip/hip_runtime.h>
#include <hip/hip_bf16.h>

typedef __attribute__((ext_vector_type(16))) __bf16 v16bf;
typedef __attribute__((ext_vector_type(8)))  float  v8f;
typedef __attribute__((ext_vector_type(4)))  unsigned int v4u;
typedef __attribute__((ext_vector_type(8)))  int v8i_t;
typedef __attribute__((ext_vector_type(4)))  int v4i_t;

#if defined(__AMDGCN__) && __has_builtin(__builtin_amdgcn_tensor_load_to_lds)
#define USE_TDM 1
#else
#define USE_TDM 0
#endif

// ---------------------------------------------------------------------------
// WMMA 16x16x32 bf16 register layouts (CDNA5 ISA 7.12.2):
//  A (16x32, MxK): lane L holds row M=L%16; element i -> K = i + (i>=8?8:0) + (L>=16?8:0)
//  B (32x16, KxN): lane L holds col N=L%16; element i -> K = i + (i>=8?8:0) + (L>=16?8:0)
//  C/D (16x16 f32): lane L -> N=L%16; vgpr r -> M = r + (L>=16?8:0)
// ---------------------------------------------------------------------------

__device__ __forceinline__ float eluf(float x) { return x > 0.f ? x : (expf(x) - 1.f); }
__device__ __forceinline__ float sigf(float x) { return 1.f / (1.f + expf(-x)); }

// TDM: DMA `n` f32 elements (n<=65535) from global to LDS byte offset.
// D# per ISA ch.8: group0 {count=1, lds_addr, global_addr, type=2},
// group1 {data_size=4B, tensor_dim0=n, tensor_dim1=1, tile_dim0=n, tile_dim1=1,
//         tensor_dim0_stride=n}. Groups 2/3 zero (<=2D).
__device__ __forceinline__ void tdm_load_f32(const float* src, unsigned lds_off, unsigned n) {
#if USE_TDM
    unsigned long long ga = (unsigned long long)(uintptr_t)src;
    v4u g0;
    g0.x = 1u;                                                   // count=1
    g0.y = lds_off;                                              // lds_addr
    g0.z = (unsigned)ga;                                         // global_addr[31:0]
    g0.w = (unsigned)((ga >> 32) & 0x01ffffffu) | (2u << 30);    // addr[56:32] | type=2
    v8i_t g1;
    g1[0] = (int)(2u << 16);                          // data_size = 4 bytes
    g1[1] = (int)((n & 0xffffu) << 16);               // tensor_dim0[15:0]  @bits63:48
    g1[2] = (int)(((n >> 16) & 0xffffu) | (1u << 16));// tensor_dim0[31:16], tensor_dim1=1
    g1[3] = (int)((n & 0xffffu) << 16);               // tile_dim0 @bits127:112
    g1[4] = 1;                                        // tile_dim1 = 1
    g1[5] = (int)n;                                   // tensor_dim0_stride[31:0]
    g1[6] = 0;
    g1[7] = 0;
    v4i_t z4 = {0, 0, 0, 0};
#if __clang_major__ >= 23
    v8i_t z8 = {0, 0, 0, 0, 0, 0, 0, 0};
    __builtin_amdgcn_tensor_load_to_lds(g0, g1, z4, z4, z8, 0);
#else
    __builtin_amdgcn_tensor_load_to_lds(g0, g1, z4, z4, 0);
#endif
#if __has_builtin(__builtin_amdgcn_s_wait_tensorcnt)
    __builtin_amdgcn_s_wait_tensorcnt(0);
#endif
#else
    (void)src; (void)lds_off; (void)n;
#endif
}

// ============================ weight packing ===============================
__global__ void pack_a_kernel(const float* __restrict__ w, __bf16* __restrict__ out,
                              int Cout, int Kdim, int nkc, long total) {
    long tid = (long)blockIdx.x * blockDim.x + threadIdx.x;
    if (tid >= total) return;
    int i    = (int)(tid & 15);
    int lane = (int)((tid >> 4) & 31);
    long rest = tid >> 9;
    int kc = (int)(rest % nkc);
    int mt = (int)(rest / nkc);
    int m = mt * 16 + (lane & 15);
    int k = kc * 32 + i + (i >= 8 ? 8 : 0) + (lane >= 16 ? 8 : 0);
    float v = (m < Cout && k < Kdim) ? w[(size_t)m * Kdim + k] : 0.f;
    out[tid] = (__bf16)v;
}

__global__ void pack_b_kernel(const float* __restrict__ w0, const float* __restrict__ w1,
                              __bf16* __restrict__ out,
                              int N, int Kdim, int Ksplit, int nkc, long total) {
    long tid = (long)blockIdx.x * blockDim.x + threadIdx.x;
    if (tid >= total) return;
    int i    = (int)(tid & 15);
    int lane = (int)((tid >> 4) & 31);
    long rest = tid >> 9;
    int kc = (int)(rest % nkc);
    int nt = (int)(rest / nkc);
    int n = nt * 16 + (lane & 15);
    int k = kc * 32 + i + (i >= 8 ? 8 : 0) + (lane >= 16 ? 8 : 0);
    float v = 0.f;
    if (n < N && k < Kdim)
        v = (k < Ksplit) ? w0[(size_t)n * Ksplit + k]
                         : w1[(size_t)n * Ksplit + (k - Ksplit)];
    out[tid] = (__bf16)v;
}

__global__ void add2_kernel(const float* a, const float* b, float* o, int n) {
    int i = blockIdx.x * blockDim.x + threadIdx.x;
    if (i < n) o[i] = a[i] + b[i];
}

__global__ void cbnorm_kernel(const float* __restrict__ cb, float* __restrict__ ccn, int total) {
    int i = blockIdx.x * blockDim.x + threadIdx.x;
    if (i >= total) return;
    const float* row = cb + (size_t)i * 128;
    float s = 0.f;
    for (int d = 0; d < 128; ++d) s += row[d] * row[d];
    ccn[i] = s;
}

// ============================ generic conv =================================
// Causal conv1d as implicit GEMM; KW/STRIDE compile-time so the im2col
// (ci,kw) = divmod(k, KW) strength-reduces (no v_mul_hi division chains).
template <int KW, int STRIDE, bool ELU>
__global__ void conv1d_wmma(const float* __restrict__ in, const __bf16* __restrict__ wp,
                            const float* __restrict__ bias, const float* __restrict__ resid,
                            float* __restrict__ out,
                            int Cin, int Cout, int Tin, int Tout) {
    constexpr int PAD = KW - STRIDE;
    const int lane = threadIdx.x;
    const int half = lane >> 4;
    const int l16  = lane & 15;
    const int t0 = blockIdx.x * 16;
    const int mt = blockIdx.y;
    const int b  = blockIdx.z;
    const int Kdim = Cin * KW;
    const int nkc  = (Kdim + 31) / 32;
    const int tbase = (t0 + l16) * STRIDE - PAD;

    v8f acc = {};
#pragma unroll 1
    for (int kc = 0; kc < nkc; ++kc) {
        size_t abase = ((size_t)(mt * nkc + kc) * 32 + lane) * 16;
        const v16bf a = *reinterpret_cast<const v16bf*>(wp + abase);
        if (kc + 1 < nkc)
            __builtin_prefetch(wp + abase + 512 * 16, 0, 1);   // global_prefetch_b8
        v16bf bv;
#pragma unroll
        for (int i = 0; i < 16; ++i) {
            int k = kc * 32 + i + (i >= 8 ? 8 : 0) + (half ? 8 : 0);
            float v = 0.f;
            if (k < Kdim) {
                int ci = k / KW;           // compile-time KW -> cheap
                int kw = k - ci * KW;
                int tin = tbase + kw;
                if (tin >= 0 && tin < Tin) {
                    v = in[((size_t)b * Cin + ci) * Tin + tin];
                    if (ELU) v = eluf(v);
                }
            }
            bv[i] = (__bf16)v;
        }
        acc = __builtin_amdgcn_wmma_f32_16x16x32_bf16(false, a, false, bv,
                                                      (short)0, acc, false, false);
    }
#pragma unroll
    for (int r = 0; r < 8; ++r) {
        int m  = r + (half ? 8 : 0);
        int co = mt * 16 + m;
        int t  = t0 + l16;
        if (co < Cout && t < Tout) {
            float v = acc[r] + bias[co];
            if (resid) v += resid[((size_t)b * Cout + co) * Tout + t];
            out[((size_t)b * Cout + co) * Tout + t] = v;
        }
    }
}

// ============================ LSTM (persistent) ============================
// One workgroup (1024 thr = 32 waves); h,c live in LDS across all 750 steps.
// sA holds [x_t ; h] (16x1024, rows 8..15 zero) directly in WMMA A-fragment
// order: sA[(kc*32+lane)*16 + i]  ->  one ds_load_b128 per K-chunk.
// x_t slab is DMA'd to LDS by the Tensor Data Mover each step.
// Weight tiles deliberately re-stream from L2 each step: the base pointer is
// laundered through inline asm per iteration so LICM cannot hoist (and spill)
// the 4KB/lane of B-fragments across the 750-step loop.
__global__ void lstm_wmma(const float* __restrict__ xin, const __bf16* __restrict__ wpk,
                          const float* __restrict__ bias, float* __restrict__ yout, int T) {
    __shared__ __align__(32) __bf16 sA[32 * 32 * 16];   // 32 K-chunks (x: 0..15, h: 16..31)
    __shared__ float sc[8 * 512];
    __shared__ float sxf[4096];                          // TDM landing pad (f32 x_t)
    const int tid  = threadIdx.x;
    const int wave = tid >> 5;
    const int lane = tid & 31;
    const int half = lane >> 4;
    const int l16  = lane & 15;

    for (int idx = tid; idx < 16384; idx += 1024) sA[idx] = (__bf16)0.f;
    for (int idx = tid; idx < 4096; idx += 1024) sc[idx] = 0.f;
    __syncthreads();

#pragma unroll 1
    for (int t = 0; t < T; ++t) {
        // defeat loop-invariant hoisting of the streamed weight tiles
        const __bf16* wv = wpk;
#if defined(__AMDGCN__)
        asm volatile("" : "+s"(wv));
#endif
#if USE_TDM
        if (wave == 0)
            tdm_load_f32(xin + (size_t)t * 4096, (unsigned)(uintptr_t)&sxf[0], 4096);
#else
        for (int idx = tid; idx < 4096; idx += 1024)
            sxf[idx] = xin[(size_t)t * 4096 + idx];
#endif
        __syncthreads();
        // convert x_t into A-fragment layout (K-chunks 0..15)
        for (int idx = tid; idx < 8192; idx += 1024) {
            int kcL = idx >> 9;
            int ln  = (idx >> 4) & 31;
            int i   = idx & 15;
            int m   = ln & 15;
            int k   = kcL * 32 + i + (i >= 8 ? 8 : 0) + (ln >= 16 ? 8 : 0);
            sA[idx] = (m < 8) ? (__bf16)sxf[m * 512 + k] : (__bf16)0.f;
        }
        __syncthreads();

        v8f acc0 = {}, acc1 = {}, acc2 = {}, acc3 = {};
#pragma unroll 1
        for (int kc = 0; kc < 32; ++kc) {
            const v16bf a = *reinterpret_cast<const v16bf*>(sA + ((size_t)kc * 32 + lane) * 16);
            size_t b0 = ((size_t)((0 * 32 + wave) * 32 + kc) * 32 + lane) * 16;
            size_t b1 = ((size_t)((1 * 32 + wave) * 32 + kc) * 32 + lane) * 16;
            size_t b2 = ((size_t)((2 * 32 + wave) * 32 + kc) * 32 + lane) * 16;
            size_t b3 = ((size_t)((3 * 32 + wave) * 32 + kc) * 32 + lane) * 16;
            __builtin_prefetch(wv + b0 + 512 * 16, 0, 1);   // global_prefetch_b8
            const v16bf w0 = *reinterpret_cast<const v16bf*>(wv + b0);
            const v16bf w1 = *reinterpret_cast<const v16bf*>(wv + b1);
            const v16bf w2 = *reinterpret_cast<const v16bf*>(wv + b2);
            const v16bf w3 = *reinterpret_cast<const v16bf*>(wv + b3);
            acc0 = __builtin_amdgcn_wmma_f32_16x16x32_bf16(false, a, false, w0, (short)0, acc0, false, false);
            acc1 = __builtin_amdgcn_wmma_f32_16x16x32_bf16(false, a, false, w1, (short)0, acc1, false, false);
            acc2 = __builtin_amdgcn_wmma_f32_16x16x32_bf16(false, a, false, w2, (short)0, acc2, false, false);
            acc3 = __builtin_amdgcn_wmma_f32_16x16x32_bf16(false, a, false, w3, (short)0, acc3, false, false);
        }
        __syncthreads();   // all reads of h fragments done before update

        const int j = (wave << 4) + l16;
#pragma unroll
        for (int r = 0; r < 8; ++r) {
            int m = r + (half ? 8 : 0);
            if (m < 8) {
                float iv = acc0[r] + bias[j];
                float fv = acc1[r] + bias[512 + j];
                float gv = acc2[r] + bias[1024 + j];
                float ov = acc3[r] + bias[1536 + j];
                float cn = sigf(fv) * sc[m * 512 + j] + sigf(iv) * tanhf(gv);
                float hn = sigf(ov) * tanhf(cn);
                sc[m * 512 + j] = cn;
                // write h back into its A-fragment slot (inverse layout map)
                int kg = 512 + j;
                int kc = kg >> 5;
                int kk = kg & 31;
                int hh = (kk >> 3) & 1;
                int ii = (kk & 7) + (kk >= 16 ? 8 : 0);
                sA[((size_t)kc * 32 + (m + (hh ? 16 : 0))) * 16 + ii] = (__bf16)hn;
                yout[(size_t)t * 4096 + m * 512 + j] = hn;
            }
        }
        __syncthreads();
    }
}

// ============================ layout shuffles ==============================
__global__ void transpose_bct_to_tbc(const float* __restrict__ in, float* __restrict__ out,
                                     int C, int T, long total) {
    long idx = (long)blockIdx.x * blockDim.x + threadIdx.x;
    if (idx >= total) return;
    long per = (long)C * T;
    int b = (int)(idx / per);
    long r = idx - (long)b * per;
    int c = (int)(r / T);
    int t = (int)(r - (long)c * T);
    out[((size_t)t * 8 + b) * C + c] = in[idx];
}

__global__ void add_tbc_kernel(const float* __restrict__ h, const float* __restrict__ y,
                               float* __restrict__ out, int C, int T, long total) {
    long idx = (long)blockIdx.x * blockDim.x + threadIdx.x;
    if (idx >= total) return;
    long per = (long)C * T;
    int b = (int)(idx / per);
    long r = idx - (long)b * per;
    int c = (int)(r / T);
    int t = (int)(r - (long)c * T);
    out[idx] = h[idx] + y[((size_t)t * 8 + b) * C + c];
}

__global__ void rvq_init_kernel(const float* __restrict__ emb, float* __restrict__ xres,
                                int T, long total) {
    long idx = (long)blockIdx.x * blockDim.x + threadIdx.x;
    if (idx >= total) return;
    long per = (long)128 * T;
    int b = (int)(idx / per);
    long r = idx - (long)b * per;
    int d = (int)(r / T);
    int t = (int)(r - (long)d * T);
    xres[((size_t)b * T + t) * 128 + d] = emb[idx];
}

// ============================ RVQ step =====================================
// 16 residual rows x 1024 codes per block. x-tile staged in A-fragment
// layout (ds_load_b128 per chunk); argmin via packed (f32bits<<32|idx)
// ds_min_u64 in LDS; then residual update.
__global__ void rvq_step(const float* __restrict__ cb, const __bf16* __restrict__ cbpk,
                         const float* __restrict__ ccn, float* __restrict__ xres,
                         short* __restrict__ codes, int q, int Nrows, int T) {
    __shared__ __align__(32) __bf16 sxA[4 * 32 * 16];
    __shared__ float  sxx[16];
    __shared__ unsigned long long best[16];
    __shared__ int sidx[16];

    const int tid  = threadIdx.x;
    const int wave = tid >> 5;
    const int lane = tid & 31;
    const int half = lane >> 4;
    const int l16  = lane & 15;
    const int n0 = blockIdx.x * 16;

    for (int idx = tid; idx < 2048; idx += 256) {
        int kc = idx >> 9;
        int ln = (idx >> 4) & 31;
        int i  = idx & 15;
        int m  = ln & 15;
        int k  = kc * 32 + i + (i >= 8 ? 8 : 0) + (ln >= 16 ? 8 : 0);
        int g  = n0 + m;
        sxA[idx] = (g < Nrows) ? (__bf16)xres[(size_t)g * 128 + k] : (__bf16)0.f;
    }
    if (tid < 16) {
        best[tid] = ~0ull;
        int g = n0 + tid;
        float s = 0.f;
        if (g < Nrows)
            for (int d = 0; d < 128; ++d) {
                float v = xres[(size_t)g * 128 + d];
                s += v * v;
            }
        sxx[tid] = s;
    }
    __syncthreads();

#pragma unroll 1
    for (int sub = 0; sub < 8; ++sub) {
        int nt = wave * 8 + sub;
        v8f acc = {};
#pragma unroll 1
        for (int kc = 0; kc < 4; ++kc) {
            const v16bf a = *reinterpret_cast<const v16bf*>(sxA + ((size_t)kc * 32 + lane) * 16);
            size_t bbase = (((size_t)(q * 64 + nt) * 4 + kc) * 32 + lane) * 16;
            const v16bf bv = *reinterpret_cast<const v16bf*>(cbpk + bbase);
            acc = __builtin_amdgcn_wmma_f32_16x16x32_bf16(false, a, false, bv,
                                                          (short)0, acc, false, false);
        }
#pragma unroll
        for (int r = 0; r < 8; ++r) {
            int m = r + (half ? 8 : 0);
            int n = nt * 16 + l16;
            float dist = fmaxf(sxx[m] - 2.f * acc[r] + ccn[q * 1024 + n], 0.f);
            unsigned long long key =
                ((unsigned long long)__float_as_uint(dist) << 32) | (unsigned)n;
            atomicMin(&best[m], key);                      // ds_min_u64
        }
    }
    __syncthreads();

    if (tid < 16) {
        int g = n0 + tid;
        int idx = (int)(best[tid] & 0xffffffffull);
        sidx[tid] = idx;
        if (g < Nrows) {
            int b = g / T, t = g - b * T;
            codes[((size_t)b * 8 + q) * T + t] = (short)idx;
        }
    }
    __syncthreads();

    for (int idx = tid; idx < 2048; idx += 256) {
        int row = idx >> 7, d = idx & 127;
        int g = n0 + row;
        if (g < Nrows)
            xres[(size_t)g * 128 + d] -= cb[((size_t)q * 1024 + sidx[row]) * 128 + d];
    }
}

// ============================ host driver ==================================
extern "C" void kernel_launch(void* const* d_in, const int* in_sizes, int n_in,
                              void* d_out, int out_size, void* d_ws, size_t ws_size,
                              hipStream_t stream) {
    (void)in_sizes; (void)n_in; (void)out_size; (void)ws_size;
    const float* x_in = (const float*)d_in[0];
    const float* b_in   = (const float*)d_in[2];
    const float* b_out  = (const float*)d_in[3];
    const float* cbooks = (const float*)d_in[4];
    const float* down_b[4] = {(const float*)d_in[5], (const float*)d_in[7],
                              (const float*)d_in[9], (const float*)d_in[11]};
    const float* down_w[4] = {(const float*)d_in[6], (const float*)d_in[8],
                              (const float*)d_in[10], (const float*)d_in[12]};
    const float* l_bhh[2] = {(const float*)d_in[13], (const float*)d_in[17]};
    const float* l_bih[2] = {(const float*)d_in[14], (const float*)d_in[18]};
    const float* l_whh[2] = {(const float*)d_in[15], (const float*)d_in[19]};
    const float* l_wih[2] = {(const float*)d_in[16], (const float*)d_in[20]};
    const float* res_b1[4] = {(const float*)d_in[21], (const float*)d_in[25],
                              (const float*)d_in[29], (const float*)d_in[33]};
    const float* res_b2[4] = {(const float*)d_in[22], (const float*)d_in[26],
                              (const float*)d_in[30], (const float*)d_in[34]};
    const float* res_w1[4] = {(const float*)d_in[23], (const float*)d_in[27],
                              (const float*)d_in[31], (const float*)d_in[35]};
    const float* res_w2[4] = {(const float*)d_in[24], (const float*)d_in[28],
                              (const float*)d_in[32], (const float*)d_in[36]};
    const float* w_in  = (const float*)d_in[37];
    const float* w_out = (const float*)d_in[38];

    const int B = 8, S = 240000, Tq = 750, Nrows = 8 * 750;
    const int strides[4] = {2, 4, 5, 8};

    size_t off = 0;
    auto alloc = [&](size_t bytes) -> void* {
        off = (off + 255) & ~(size_t)255;
        void* p = (char*)d_ws + off;
        off += bytes;
        return p;
    };
    float* BUF0 = (float*)alloc((size_t)61440000 * 4);
    float* BUF1 = (float*)alloc((size_t)61440000 * 4);
    float* BUF2 = (float*)alloc((size_t)30720000 * 4);
    float* xT   = (float*)alloc((size_t)3072000 * 4);
    float* yL0  = (float*)alloc((size_t)3072000 * 4);
    float* yL1  = (float*)alloc((size_t)3072000 * 4);
    float* hfin = (float*)alloc((size_t)3072000 * 4);
    float* emb  = (float*)alloc((size_t)768000 * 4);
    float* xres = (float*)alloc((size_t)768000 * 4);
    float* bcomb = (float*)alloc((size_t)2 * 2048 * 4);
    float* ccn   = (float*)alloc((size_t)8192 * 4);

    auto packA = [&](const float* w, int Cout, int Cin, int Kw) -> const __bf16* {
        int Kdim = Cin * Kw;
        int nmt = (Cout + 15) / 16, nkc = (Kdim + 31) / 32;
        long n = (long)nmt * nkc * 512;
        __bf16* p = (__bf16*)alloc((size_t)n * 2);
        pack_a_kernel<<<(int)((n + 255) / 256), 256, 0, stream>>>(w, p, Cout, Kdim, nkc, n);
        return p;
    };
    auto packB = [&](const float* w0, const float* w1, int N, int Kdim, int Ksplit)
        -> const __bf16* {
        int nnt = (N + 15) / 16, nkc = (Kdim + 31) / 32;
        long n = (long)nnt * nkc * 512;
        __bf16* p = (__bf16*)alloc((size_t)n * 2);
        pack_b_kernel<<<(int)((n + 255) / 256), 256, 0, stream>>>(w0, w1, p, N, Kdim, Ksplit, nkc, n);
        return p;
    };

    const __bf16* pw_in = packA(w_in, 32, 1, 7);
    const __bf16* pw1[4]; const __bf16* pw2[4]; const __bf16* pdw[4];
    {
        int c = 32;
        for (int i = 0; i < 4; ++i) {
            pw1[i] = packA(res_w1[i], c / 2, c, 3);
            pw2[i] = packA(res_w2[i], c, c / 2, 1);
            pdw[i] = packA(down_w[i], 2 * c, c, 2 * strides[i]);
            c *= 2;
        }
    }
    const __bf16* pw_out = packA(w_out, 128, 512, 7);
    const __bf16* pl[2];
    for (int l = 0; l < 2; ++l) {
        pl[l] = packB(l_wih[l], l_whh[l], 2048, 1024, 512);
        add2_kernel<<<8, 256, 0, stream>>>(l_bih[l], l_bhh[l], bcomb + l * 2048, 2048);
    }
    const __bf16* pcb = packB(cbooks, nullptr, 8192, 128, 128);
    cbnorm_kernel<<<32, 256, 0, stream>>>(cbooks, ccn, 8192);

    // ---- encoder convs (templated KW/STRIDE dispatch) ----
    auto conv = [&](const float* in, const __bf16* wp, const float* bias,
                    const float* resid, float* out, int Cin, int Cout, int Kw,
                    int stride, int Tin, int elu) -> int {
        int Tout = Tin / stride;
        dim3 grid((Tout + 15) / 16, (Cout + 15) / 16, B);
#define LAUNCH_CONV(KW_, S_, E_) \
        conv1d_wmma<KW_, S_, E_><<<grid, 32, 0, stream>>>(in, wp, bias, resid, out, \
                                                          Cin, Cout, Tin, Tout)
        if      (Kw == 7  && stride == 1 && !elu) LAUNCH_CONV(7, 1, false);
        else if (Kw == 7  && stride == 1)         LAUNCH_CONV(7, 1, true);
        else if (Kw == 3)                         LAUNCH_CONV(3, 1, true);
        else if (Kw == 1)                         LAUNCH_CONV(1, 1, true);
        else if (Kw == 4)                         LAUNCH_CONV(4, 2, true);
        else if (Kw == 8)                         LAUNCH_CONV(8, 4, true);
        else if (Kw == 10)                        LAUNCH_CONV(10, 5, true);
        else if (Kw == 16)                        LAUNCH_CONV(16, 8, true);
#undef LAUNCH_CONV
        return Tout;
    };

    conv(x_in, pw_in, b_in, nullptr, BUF0, 1, 32, 7, 1, S, 0);
    int T = S, c = 32;
    for (int i = 0; i < 4; ++i) {
        conv(BUF0, pw1[i], res_b1[i], nullptr, BUF2, c, c / 2, 3, 1, T, 1);
        conv(BUF2, pw2[i], res_b2[i], BUF0, BUF1, c / 2, c, 1, 1, T, 1);
        T = conv(BUF1, pdw[i], down_b[i], nullptr, BUF0, c, 2 * c, 2 * strides[i],
                 strides[i], T, 1);
        c *= 2;
    }

    long nt = (long)B * 512 * Tq;
    transpose_bct_to_tbc<<<(int)((nt + 255) / 256), 256, 0, stream>>>(BUF0, xT, 512, Tq, nt);
    lstm_wmma<<<1, 1024, 0, stream>>>(xT, pl[0], bcomb, yL0, Tq);
    lstm_wmma<<<1, 1024, 0, stream>>>(yL0, pl[1], bcomb + 2048, yL1, Tq);
    add_tbc_kernel<<<(int)((nt + 255) / 256), 256, 0, stream>>>(BUF0, yL1, hfin, 512, Tq, nt);
    conv(hfin, pw_out, b_out, nullptr, emb, 512, 128, 7, 1, Tq, 1);

    long ne = (long)B * 128 * Tq;
    rvq_init_kernel<<<(int)((ne + 255) / 256), 256, 0, stream>>>(emb, xres, Tq, ne);
    short* codes = (short*)d_out;
    for (int q = 0; q < 8; ++q)
        rvq_step<<<(Nrows + 15) / 16, 256, 0, stream>>>(cbooks, pcb, ccn, xres, codes,
                                                        q, Nrows, Tq);
}
